// TBRNN_39290360824508
// MI455X (gfx1250) — matrix operand order repaired and checked by reading, not compiled, not run on hardware
//
#include <hip/hip_runtime.h>
#include <hip/hip_bf16.h>
#include <math.h>

// Problem constants: B=16, T=512, I=8, H=256
// Dominant op per step: rec = R * W^T, M=16, N=256, K=65536 (bf16 WMMA, f32 acc)

typedef __attribute__((ext_vector_type(16))) __bf16       v16bf;
typedef __attribute__((ext_vector_type(8)))  float        v8f;
typedef __attribute__((ext_vector_type(8)))  unsigned int v8u;

__device__ __forceinline__ unsigned int bf16_rne(float f) {
    unsigned int u = __builtin_bit_cast(unsigned int, f);
    return (u + 0x7FFFu + ((u >> 16) & 1u)) >> 16;   // round-to-nearest-even
}
__device__ __forceinline__ unsigned int pack_bf16(float lo, float hi) {
    return bf16_rne(lo) | (bf16_rne(hi) << 16);      // [15:0]=even K, [31:16]=odd K
}

__device__ __forceinline__ v16bf make_amat(const uint4 a0, const uint4 a1) {
    v8u au;
    au[0] = a0.x; au[1] = a0.y; au[2] = a0.z; au[3] = a0.w;
    au[4] = a1.x; au[5] = a1.y; au[6] = a1.z; au[7] = a1.w;
    return __builtin_bit_cast(v16bf, au);
}

__device__ __forceinline__ v16bf make_bmat(const uint4 r0, const uint4 r1,
                                           const unsigned int rj2) {
    unsigned int b0, b1, b2, b3, b4, b5, b6, b7;
    asm("v_pk_mul_bf16 %0, %1, %2" : "=v"(b0) : "v"(r0.x), "v"(rj2));
    asm("v_pk_mul_bf16 %0, %1, %2" : "=v"(b1) : "v"(r0.y), "v"(rj2));
    asm("v_pk_mul_bf16 %0, %1, %2" : "=v"(b2) : "v"(r0.z), "v"(rj2));
    asm("v_pk_mul_bf16 %0, %1, %2" : "=v"(b3) : "v"(r0.w), "v"(rj2));
    asm("v_pk_mul_bf16 %0, %1, %2" : "=v"(b4) : "v"(r1.x), "v"(rj2));
    asm("v_pk_mul_bf16 %0, %1, %2" : "=v"(b5) : "v"(r1.y), "v"(rj2));
    asm("v_pk_mul_bf16 %0, %1, %2" : "=v"(b6) : "v"(r1.z), "v"(rj2));
    asm("v_pk_mul_bf16 %0, %1, %2" : "=v"(b7) : "v"(r1.w), "v"(rj2));
    v8u bu;
    bu[0] = b0; bu[1] = b1; bu[2] = b2; bu[3] = b3;
    bu[4] = b4; bu[5] = b5; bu[6] = b6; bu[7] = b7;
    return __builtin_bit_cast(v16bf, bu);
}

// ---------------------------------------------------------------------------
// Prologue: swizzle w_hh (fp32 [256,65536]) into the exact WMMA-A per-lane
// register image, bf16-packed: Wsw[((kstep*16 + itile)*32 + lane)*8 + i].
// A 16x32 bf16 layout (ISA 7.12.2): lane m=L%16,h=L/16; v[i<4]: K=8h+2i(,+1);
// v[4+i]: K=16+8h+2i(,+1). Also copies x0 into the ping x-buffer.
// ---------------------------------------------------------------------------
__global__ __launch_bounds__(1024)
void tbrnn_prologue(const float* __restrict__ w_hh, const float* __restrict__ x0,
                    unsigned int* __restrict__ Wsw, float* __restrict__ xbuf0) {
    const unsigned int id    = blockIdx.x * 1024u + threadIdx.x;   // 0 .. 8388607
    const unsigned int i     = id & 7u;
    const unsigned int lane  = (id >> 3) & 31u;
    const unsigned int itile = (id >> 8) & 15u;
    const unsigned int kstep = id >> 12;                           // 0 .. 2047
    const unsigned int m  = lane & 15u;
    const unsigned int hh = lane >> 4;
    const unsigned int c  = (i < 4u) ? (8u * hh + 2u * i) : (16u + 8u * hh + 2u * (i - 4u));
    const unsigned int row = itile * 16u + m;                      // output channel i
    const unsigned int kk  = kstep * 32u + c;                      // kk = j*256 + k
    const float* src = w_hh + (size_t)row * 65536u + kk;
    Wsw[id] = pack_bf16(src[0], src[1]);
    if (id < 4096u) xbuf0[id] = x0[id];
}

// ---------------------------------------------------------------------------
// One recurrence step. Grid: 16 WGs (one per 16-wide i-tile) x 1024 threads
// (32 waves). Wave w handles ksteps [w*64, w*64+64) of the K=65536 contraction.
// Traversal is slot-major (kstep = jg*8 + s): the B-side r-pairs depend only
// on s, so each slot's two b128s are loaded from LDS once and reused for 8
// WMMAs (8x less LDS traffic). A tiles are register double-buffered so loads
// stay in flight across the WMMA and no hazard nops are needed.
// ---------------------------------------------------------------------------
__global__ __launch_bounds__(1024)
void tbrnn_step(int t, const float* __restrict__ xr, float* __restrict__ xw,
                const unsigned int* __restrict__ Wsw,
                const float* __restrict__ u, const float* __restrict__ noise,
                const float* __restrict__ w_in_w, const float* __restrict__ w_in_b,
                float* __restrict__ traj) {
    __shared__ unsigned int rpair[16 * 132];   // r=tanh(x) as packed bf16 pairs, padded stride
    __shared__ float accs[256];                // 16x16 f32 reduction tile

    const unsigned int tid = threadIdx.x;
    // Fill r pairs: 16 batch rows x 128 pairs = 2048 entries, 2 per thread.
    for (unsigned int s = tid; s < 2048u; s += 1024u) {
        const unsigned int b = s >> 7, p = s & 127u;
        const float* xp = xr + b * 256u + 2u * p;
        rpair[b * 132u + p] = pack_bf16(tanhf(xp[0]), tanhf(xp[1]));
    }
    if (tid < 256u) accs[tid] = 0.0f;
    __syncthreads();

    const unsigned int lane  = tid & 31u;
    const unsigned int wave  = tid >> 5;       // 0..31
    const unsigned int hh    = lane >> 4;
    const unsigned int bcol  = lane & 15u;     // batch column of B/D
    const unsigned int itile = blockIdx.x;     // 0..15
    const unsigned int rbase = bcol * 132u;
    const int kbase = (int)wave * 64;

    // All 8 r[b,j] scalars this wave needs, via one LDS b128:
    // j in [wave*8, wave*8+8) -> pairs rpair[rbase + wave*4 .. +3].
    const uint4 rjs = *(const uint4*)&rpair[rbase + wave * 4u];

    // Precompute the 8 rj2 splats (static selects -> 8 VGPRs).
    unsigned int rj2v[8];
#pragma unroll
    for (int jg = 0; jg < 8; ++jg) {
        unsigned int v;
        switch (jg >> 1) {
            case 0:  v = rjs.x; break;
            case 1:  v = rjs.y; break;
            case 2:  v = rjs.z; break;
            default: v = rjs.w; break;
        }
        const unsigned int sel = (jg & 1) ? (v >> 16) : (v & 0xFFFFu);
        rj2v[jg] = sel | (sel << 16);
    }

    // Tile base pointers; per-kstep strides become immediate offsets.
    const uint4* __restrict__ abase =
        (const uint4*)(Wsw + (((unsigned int)kbase * 16u + itile) << 8) + (lane << 3));
    const uint4* __restrict__ rb = (const uint4*)&rpair[rbase + (hh << 2)];

    uint4 a0[2], a1[2];
    a0[0] = abase[0];                  // kstep = jg*8 + s = 0
    a1[0] = abase[1];
    uint4 rc0 = rb[0], rc1 = rb[2];    // slot s = 0
    uint4 rn0 = rc0, rn1 = rc1;

    v8f acc = {0.f, 0.f, 0.f, 0.f, 0.f, 0.f, 0.f, 0.f};

    // idx = s*8 + jg ; kstep = jg*8 + s. One unrolled body == one slot s.
#pragma unroll 8
    for (int idx = 0; idx < 64; ++idx) {
        const int jg  = idx & 7;               // static per unrolled instance
        const int s   = idx >> 3;              // slot (loop counter)
        const int cur = idx & 1;               // static (8 even)
        const int nxt = cur ^ 1;
        const int idxp = (idx + 1) & 63;       // wrap keeps pipeline branchless
        // ---- A prefetch for the next instance: offset = (jg'*8 + s')*1024 u4 ----
        const unsigned int koff = (unsigned int)(((idxp & 7) << 3) + (idxp >> 3)) * 1024u;
        a0[nxt] = abase[koff];
        a1[nxt] = abase[koff + 1u];
        // ---- entering a slot: prefetch next slot's r-pairs (once per 8 WMMAs) ----
        if (jg == 0) {
            const unsigned int snx = (unsigned int)(s + 1) & 7u;
            rn0 = rb[snx * 4u];
            rn1 = rb[snx * 4u + 2u];
        }
        // ---- consume ----
        const v16bf Bm = make_bmat(rc0, rc1, rj2v[jg]);
        const v16bf Am = make_amat(a0[cur], a1[cur]);
        acc = __builtin_amdgcn_wmma_f32_16x16x32_bf16(false, Am, false, Bm,
                                                      (short)0, acc, false, false);
        if (jg == 7) { rc0 = rn0; rc1 = rn1; }  // rotate slot buffer
    }

    // D layout: VGPR v, lane -> (M = v + 8h, N = bcol); reduce across 32 waves.
#pragma unroll
    for (int v = 0; v < 8; ++v)
        atomicAdd(&accs[(unsigned int)((v + (int)(hh << 3)) << 4) + bcol], acc[v]);
    __syncthreads();

    // Elementwise Euler update for this WG's 16 columns.
    if (tid < 256u) {
        const unsigned int b    = tid & 15u;
        const unsigned int iloc = tid >> 4;
        const unsigned int ic   = itile * 16u + iloc;
        const float rec = accs[iloc * 16u + b];
        const float xo  = xr[b * 256u + ic];
        float up = w_in_b[ic];
        const float* ur = u + (size_t)b * 4096u + (size_t)t * 8u;   // u[b,t,:]
        const float* wr = w_in_w + ic * 8u;                         // w_in_w[ic,:]
#pragma unroll
        for (int q = 0; q < 8; ++q) up += ur[q] * wr[q];
        const float nt = noise[(size_t)t * 4096u + b * 256u + ic];
        const float xn = xo + 0.05f * nt + 0.2f * (-xo + rec + up);
        xw[b * 256u + ic] = xn;
        traj[(size_t)b * 131072u + (size_t)t * 256u + ic] = xn;
    }
}

// ---------------------------------------------------------------------------
// Output projection: one wave per (b,t); lane covers 8 h-values, butterfly
// reduce 8 partial dots across the wave (wave32), lane 0 writes 8 outputs.
// ---------------------------------------------------------------------------
__global__ __launch_bounds__(256)
void tbrnn_out(const float* __restrict__ traj, const float* __restrict__ w_out_w,
               const float* __restrict__ w_out_b, float* __restrict__ out) {
    const unsigned int gtid = blockIdx.x * 256u + threadIdx.x;
    const unsigned int wid  = gtid >> 5;            // 0..8191
    const unsigned int lane = gtid & 31u;
    const unsigned int b = wid >> 9, t = wid & 511u;
    const float* tr = traj + (size_t)b * 131072u + (size_t)t * 256u + lane * 8u;
    float v[8];
#pragma unroll
    for (int c = 0; c < 8; ++c) v[c] = tanhf(tr[c]);
    float acc[8];
#pragma unroll
    for (int o = 0; o < 8; ++o) {
        const float* wr = w_out_w + (unsigned int)o * 256u + lane * 8u;
        float s = 0.f;
#pragma unroll
        for (int c = 0; c < 8; ++c) s += v[c] * wr[c];
        acc[o] = s;
    }
#pragma unroll
    for (int m = 16; m >= 1; m >>= 1) {
#pragma unroll
        for (int o = 0; o < 8; ++o) acc[o] += __shfl_xor(acc[o], m, 32);
    }
    if (lane == 0) {
        float* op = out + (size_t)b * 4096u + (size_t)t * 8u;
#pragma unroll
        for (int o = 0; o < 8; ++o) op[o] = acc[o] + w_out_b[o];
    }
}

__global__ void tbrnn_xlast(const float* __restrict__ traj, float* __restrict__ xlast) {
    const unsigned int id = blockIdx.x * 256u + threadIdx.x;
    if (id < 4096u) {
        const unsigned int b = id >> 8, hc = id & 255u;
        xlast[id] = traj[(size_t)b * 131072u + 511u * 256u + hc];
    }
}

// ---------------------------------------------------------------------------
extern "C" void kernel_launch(void* const* d_in, const int* in_sizes, int n_in,
                              void* d_out, int out_size, void* d_ws, size_t ws_size,
                              hipStream_t stream) {
    (void)in_sizes; (void)n_in; (void)out_size; (void)ws_size;
    const float* u       = (const float*)d_in[0];  // [16,512,8]
    const float* x0      = (const float*)d_in[1];  // [16,256]
    const float* noise   = (const float*)d_in[2];  // [512,16,256]
    const float* w_hh    = (const float*)d_in[3];  // [256,256,256]
    const float* w_in_w  = (const float*)d_in[4];  // [256,8]
    const float* w_in_b  = (const float*)d_in[5];  // [256]
    const float* w_out_w = (const float*)d_in[6];  // [8,256]
    const float* w_out_b = (const float*)d_in[7];  // [8]

    float* out     = (float*)d_out;
    float* out_y   = out;             // output [16,512,8]  = 65536 f32
    float* out_xl  = out + 65536;     // x_last [16,256]    = 4096  f32
    float* out_tr  = out + 69632;     // traj   [16,512,256]= 2097152 f32

    unsigned int* Wsw = (unsigned int*)d_ws;        // 8388608 dwords = 33.5 MB (L2-resident)
    float* xb = (float*)(Wsw + 8388608);            // ping/pong x: 2 x 4096 f32

    tbrnn_prologue<<<8192, 1024, 0, stream>>>(w_hh, x0, Wsw, xb);

    for (int t = 0; t < 512; ++t) {
        const float* xr = xb + (size_t)(t & 1) * 4096u;
        float*       xw = xb + (size_t)((t & 1) ^ 1) * 4096u;
        tbrnn_step<<<16, 1024, 0, stream>>>(t, xr, xw, Wsw, u, noise,
                                            w_in_w, w_in_b, out_tr);
    }

    tbrnn_out<<<1024, 256, 0, stream>>>(out_tr, w_out_w, w_out_b, out_y);
    tbrnn_xlast<<<16, 256, 0, stream>>>(out_tr, out_xl);
}
